// GRU_66984309948762
// MI455X (gfx1250) — compile-verified, matrix operand order checked by reading, not run on hardware
//
#include <hip/hip_runtime.h>
#include <hip/hip_bf16.h>
#include <stdint.h>

typedef __attribute__((ext_vector_type(16))) _Float16 v16h;
typedef __attribute__((ext_vector_type(8)))  float    v8f;

#define B_SZ     256
#define T_SZ     250
#define NSTK     500
#define HID      512
#define G3       1536   // 3*HID
#define MROWS    (B_SZ*T_SZ)          // 64000
#define GI_BYTES ((size_t)MROWS*G3*2) // 196,608,000
#define XH_BYTES ((size_t)MROWS*HID*2)// 65,536,000
#define WMAT_ELE (G3*HID)             // 786432 per matrix

__device__ inline v8f wmma16(v16h a, v16h b, v8f c) {
    return __builtin_amdgcn_wmma_f32_16x16x32_f16(
        false, a, false, b, (short)0, c, false, false);
}

// A/B fragment loader, row-major source with leading dim `ld` (f16).
// ISA 16-bit A 16x32 layout: lanes 0-15 -> rows, elems 0..7 = K k0..k0+7,
// elems 8..15 = K k0+16..k0+23; lanes 16-31 use K+8 / K+24.
__device__ inline v16h load_frag(const _Float16* __restrict__ base,
                                 int row0, int ld, int k0, int lane) {
    int r   = row0 + (lane & 15);
    int klo = k0 + ((lane & 16) >> 1);          // +8 for upper lane half
    const _Float16* p = base + (size_t)r * ld + klo;
    union { uint4 u[2]; v16h h; } f;
    f.u[0] = *(const uint4*)(p);
    f.u[1] = *(const uint4*)(p + 16);
    return f.h;
}

__device__ inline float sigm(float x) { return 1.0f / (1.0f + __expf(-x)); }
__device__ inline float tanh_fast(float x) { return 2.0f / (1.0f + __expf(-2.0f * x)) - 1.0f; }

// async global -> LDS copy of 16 bytes (tracked by ASYNCcnt)
__device__ inline void async_cp_b128(uint32_t lds_addr, const void* gaddr) {
    asm volatile("global_load_async_to_lds_b128 %0, %1, off"
                 :: "v"(lds_addr), "v"(gaddr) : "memory");
}
__device__ inline void wait_async0() {
    asm volatile("s_wait_asynccnt 0" ::: "memory");
}

// ---------------- f32 -> f16 convert with K padding -------------------------
__global__ void cvt_pad(const float* __restrict__ src, _Float16* __restrict__ dst,
                        int M, int Kin, int Kpad) {
    int64_t n = (int64_t)M * Kpad;
    for (int64_t i = (int64_t)blockIdx.x * blockDim.x + threadIdx.x; i < n;
         i += (int64_t)gridDim.x * blockDim.x) {
        int r = (int)(i / Kpad), k = (int)(i % Kpad);
        dst[i] = (k < Kin) ? (_Float16)src[(int64_t)r * Kin + k] : (_Float16)0.0f;
    }
}

// ---------------- gi = A(MxK) * W(NxK)^T + bias, f16 out --------------------
// one wave computes a 32(M) x 64(N) strip; K = 512 (16 wmma k-steps)
// per k-step: 12 b128 loads, 8 wmma
__global__ __launch_bounds__(256)
void gemm_gi(const _Float16* __restrict__ A, const _Float16* __restrict__ W,
             const float* __restrict__ bias, _Float16* __restrict__ C,
             int M, int N, int K) {
    int wid  = (blockIdx.x * blockDim.x + threadIdx.x) >> 5;
    int lane = threadIdx.x & 31;
    int ngrp = N >> 6;                       // 64-col groups
    int mt   = wid / ngrp;
    int ng   = wid % ngrp;
    if (mt >= (M >> 5)) return;
    int m0 = mt << 5, n0 = ng << 6;

    v8f zero = {};
    v8f acc0[4] = {zero, zero, zero, zero};  // rows m0..m0+15
    v8f acc1[4] = {zero, zero, zero, zero};  // rows m0+16..m0+31
    for (int k0 = 0; k0 < K; k0 += 32) {
        v16h a0 = load_frag(A, m0,      K, k0, lane);
        v16h a1 = load_frag(A, m0 + 16, K, k0, lane);
        #pragma unroll
        for (int j = 0; j < 4; ++j) {
            v16h b = load_frag(W, n0 + 16 * j, K, k0, lane);
            acc0[j] = wmma16(a0, b, acc0[j]);
            acc1[j] = wmma16(a1, b, acc1[j]);
        }
    }
    // C/D layout: N = lane%16, M = 8*(lane/16) + e
    int rb = ((lane >> 4) << 3);
    #pragma unroll
    for (int j = 0; j < 4; ++j) {
        int c   = n0 + 16 * j + (lane & 15);
        float bi = bias[c];
        #pragma unroll
        for (int e = 0; e < 8; ++e) {
            C[(size_t)(m0 + rb + e)      * N + c] = (_Float16)(acc0[j][e] + bi);
            C[(size_t)(m0 + 16 + rb + e) * N + c] = (_Float16)(acc1[j][e] + bi);
        }
    }
}

// ---------------- GRU recurrence scan ---------------------------------------
// 16 blocks x 1024 threads (32 waves); block owns 16 batch rows.
// h (16x512 f16) double-buffered in LDS; gi gate slab for the *next* timestep
// is staged into LDS with async global->LDS b128 copies, overlapped with the
// WMMA compute of the current step. Wave w owns gate-triple cols [w*16,w*16+16).
__global__ __launch_bounds__(1024)
void gru_scan(const _Float16* __restrict__ gi, const _Float16* __restrict__ whh,
              const float* __restrict__ bhh, _Float16* __restrict__ h_out,
              float* __restrict__ h_final) {
    __shared__ _Float16 hbuf[2][16 * HID];   // 32 KB
    __shared__ _Float16 gist[2][16 * G3];    // 96 KB gi staging (double buffer)
    int tid  = threadIdx.x;
    int wave = tid >> 5;                     // 0..31
    int lane = tid & 31;
    int r0   = blockIdx.x << 4;              // first batch row of this block

    for (int i = tid; i < 16 * HID; i += blockDim.x) hbuf[0][i] = (_Float16)0.0f;

    // stage gi for t=0 into buffer 0: 16 rows x 3072B = 3072 x 16B chunks
    {
        uint32_t base = (uint32_t)(uintptr_t)&gist[0][0];
        for (int c = tid; c < 3072; c += 1024) {
            int pr = c / 192, pc = c % 192;
            async_cp_b128(base + (uint32_t)(pr * G3 + pc * 8) * 2,
                          gi + (size_t)((r0 + pr) * T_SZ) * G3 + pc * 8);
        }
    }
    wait_async0();
    __syncthreads();

    int   jr  = wave << 4;                   // r-gate col base; z:+512, n:+1024
    int   j   = jr + (lane & 15);            // hidden unit index, 0..511
    int   rb  = (lane >> 4) << 3;
    float bhr = bhh[j], bhz = bhh[j + 512], bhn = bhh[j + 1024];

    int cur = 0;
    v8f zero = {};
    #pragma unroll 1
    for (int t = 0; t < T_SZ; ++t) {
        const _Float16* hc = &hbuf[cur][0];
        _Float16*       hn = &hbuf[cur ^ 1][0];
        const _Float16* gr = &gist[cur][0];

        // kick off async staging of gi for t+1 into the other buffer
        if (t + 1 < T_SZ) {
            uint32_t base = (uint32_t)(uintptr_t)&gist[cur ^ 1][0];
            for (int c = tid; c < 3072; c += 1024) {
                int pr = c / 192, pc = c % 192;
                async_cp_b128(base + (uint32_t)(pr * G3 + pc * 8) * 2,
                              gi + (size_t)((r0 + pr) * T_SZ + t + 1) * G3 + pc * 8);
            }
        }

        v8f ar = zero, az = zero, an = zero;
        for (int k0 = 0; k0 < HID; k0 += 32) {
            v16h a  = load_frag(hc,  0,         HID, k0, lane);
            v16h br = load_frag(whh, jr,        HID, k0, lane);
            v16h bz = load_frag(whh, jr + 512,  HID, k0, lane);
            v16h bn = load_frag(whh, jr + 1024, HID, k0, lane);
            ar = wmma16(a, br, ar);
            az = wmma16(a, bz, az);
            an = wmma16(a, bn, an);
        }
        #pragma unroll
        for (int e = 0; e < 8; ++e) {
            int r = rb + e;                  // local batch row 0..15
            int b = r0 + r;
            float ir  = (float)gr[r * G3 + j];
            float iz  = (float)gr[r * G3 + j + 512];
            float in_ = (float)gr[r * G3 + j + 1024];
            float rr = sigm(ir + ar[e] + bhr);
            float zz = sigm(iz + az[e] + bhz);
            float nn = tanh_fast(in_ + rr * (an[e] + bhn));
            float hp = (float)hc[r * HID + j];
            float hv = (1.0f - zz) * nn + zz * hp;
            hn[r * HID + j] = (_Float16)hv;
            if (h_out)  h_out[(size_t)(b * T_SZ + t) * HID + j] = (_Float16)hv;
            if (h_final && t == T_SZ - 1) h_final[(size_t)b * HID + j] = hv;
        }
        wait_async0();                       // staged gi for t+1 landed
        __syncthreads();
        cur ^= 1;
    }
}

// ---------------- FC head + SiLU + softmax + rebalance ----------------------
__global__ __launch_bounds__(512)
void fc_head(const float* __restrict__ hfin, const float* __restrict__ fcw,
             const float* __restrict__ fcb, float* __restrict__ out) {
    int row = blockIdx.x, tid = threadIdx.x;
    __shared__ float hrow[HID];
    __shared__ float red[512];
    hrow[tid] = hfin[(size_t)row * HID + tid];
    __syncthreads();

    float logit = 0.0f;
    bool active = (tid < NSTK);
    if (active) {
        const float* wr = fcw + (size_t)tid * HID;
        float acc = fcb[tid];
        for (int k = 0; k < HID; ++k) acc = fmaf(hrow[k], wr[k], acc);
        logit = acc * sigm(acc);             // SiLU
    }
    // softmax: max-reduce
    red[tid] = active ? logit : -3.4e38f;
    __syncthreads();
    for (int s = 256; s > 0; s >>= 1) {
        if (tid < s) red[tid] = fmaxf(red[tid], red[tid + s]);
        __syncthreads();
    }
    float mx = red[0];
    __syncthreads();
    float ex = active ? __expf(logit - mx) : 0.0f;
    red[tid] = ex;
    __syncthreads();
    for (int s = 256; s > 0; s >>= 1) {
        if (tid < s) red[tid] += red[tid + s];
        __syncthreads();
    }
    float w = ex / red[0];
    __syncthreads();

    for (int it = 0; it < 30; ++it) {
        float wc = fminf(fmaxf(w, 0.0f), 0.1f);
        red[tid] = active ? (w - wc) : 0.0f;
        __syncthreads();
        for (int s = 256; s > 0; s >>= 1) {
            if (tid < s) red[tid] += red[tid + s];
            __syncthreads();
        }
        float leftover = red[0];
        __syncthreads();
        float nom = (wc != 0.1f) ? wc : 0.0f;
        red[tid] = active ? nom : 0.0f;
        __syncthreads();
        for (int s = 256; s > 0; s >>= 1) {
            if (tid < s) red[tid] += red[tid + s];
            __syncthreads();
        }
        float snom = red[0];
        __syncthreads();
        w = wc + leftover * nom / snom;
    }
    if (active) out[(size_t)row * NSTK + tid] = w;
}

extern "C" void kernel_launch(void* const* d_in, const int* in_sizes, int n_in,
                              void* d_out, int out_size, void* d_ws, size_t ws_size,
                              hipStream_t stream) {
    const float* x     = (const float*)d_in[0];
    const float* wih0  = (const float*)d_in[1];
    const float* whh0  = (const float*)d_in[2];
    const float* bih0  = (const float*)d_in[3];
    const float* bhh0  = (const float*)d_in[4];
    const float* wih1  = (const float*)d_in[5];
    const float* whh1  = (const float*)d_in[6];
    const float* bih1  = (const float*)d_in[7];
    const float* bhh1  = (const float*)d_in[8];
    const float* fcw   = (const float*)d_in[9];
    const float* fcb   = (const float*)d_in[10];

    char* ws = (char*)d_ws;
    _Float16* gi   = (_Float16*)(ws);
    _Float16* xh   = (_Float16*)(ws + GI_BYTES);           // x_f16, reused as h0
    _Float16* w0i  = (_Float16*)(ws + GI_BYTES + XH_BYTES);
    _Float16* w0h  = w0i + WMAT_ELE;
    _Float16* w1i  = w0h + WMAT_ELE;
    _Float16* w1h  = w1i + WMAT_ELE;
    float*    hfin = (float*)(ws + GI_BYTES + XH_BYTES + (size_t)4 * WMAT_ELE * 2);

    // 1) convert inputs/weights to f16, pad K 500->512 where needed
    cvt_pad<<<4096, 256, 0, stream>>>(x,    xh,  MROWS, NSTK, HID);
    cvt_pad<<<512,  256, 0, stream>>>(wih0, w0i, G3,    NSTK, HID);
    cvt_pad<<<512,  256, 0, stream>>>(whh0, w0h, G3,    HID,  HID);
    cvt_pad<<<512,  256, 0, stream>>>(wih1, w1i, G3,    HID,  HID);
    cvt_pad<<<512,  256, 0, stream>>>(whh1, w1h, G3,    HID,  HID);

    const int gemm_blocks = (MROWS / 32) * (G3 / 64) / 8;  // 6000

    // 2) layer 0: gi = x @ w_ih0^T + b_ih0 ; scan (writes h0 over xh)
    gemm_gi<<<gemm_blocks, 256, 0, stream>>>(xh, w0i, bih0, gi, MROWS, G3, HID);
    gru_scan<<<16, 1024, 0, stream>>>(gi, w0h, bhh0, xh, nullptr);

    // 3) layer 1: gi = h0 @ w_ih1^T + b_ih1 ; scan (final h only)
    gemm_gi<<<gemm_blocks, 256, 0, stream>>>(xh, w1i, bih1, gi, MROWS, G3, HID);
    gru_scan<<<16, 1024, 0, stream>>>(gi, w1h, bhh1, nullptr, hfin);

    // 4) FC + SiLU + softmax + 30x rebalance
    fc_head<<<B_SZ, 512, 0, stream>>>(hfin, fcw, fcb, (float*)d_out);
}
